// LayerEmbedder_75376676045031
// MI455X (gfx1250) — matrix operand order repaired and checked by reading, not compiled, not run on hardware
//
#include <hip/hip_runtime.h>
#include <hip/hip_bf16.h>
#include <stdint.h>

#define HID 128
#define NL  3
#define PAD 8

typedef _Float16 h8  __attribute__((ext_vector_type(8)));
typedef _Float16 h16 __attribute__((ext_vector_type(16)));
typedef float    f8  __attribute__((ext_vector_type(8)));

// ---------------------------------------------------------------------------
// CDNA5 WMMA fragment loaders (wave32, V_WMMA_F32_16X16X32_F16)
//
// A (16x32 f16): lanes 0-15 row M=lane hold K kb+0..15 contiguous;
//                lanes 16-31 row M=lane-16 hold K kb+16..31 contiguous.
// B (32x16 f16): column N = lane&15; lanes 0-15 hold K kb+0..15 contiguous,
//                lanes 16-31 hold K kb+16..31 contiguous.
// C/D (16x16 f32, 8 VGPRs): element (M = v + 8*(lane>>4), N = lane&15).
// ---------------------------------------------------------------------------
__device__ __forceinline__ h16 load_a_frag(const _Float16* rowk, int hi) {
  h8 lo = *(const h8*)(rowk + 16 * hi);
  h8 hh = *(const h8*)(rowk + 16 * hi + 8);
  return __builtin_shufflevector(lo, hh, 0, 1, 2, 3, 4, 5, 6, 7,
                                 8, 9, 10, 11, 12, 13, 14, 15);
}

// One wave computes a 16-row x 128-col f32 tile: acc += sA[16xK] * Wt^T.
// sA: LDS tile (row stride lda halves). Wt: f16, transposed weights [128][K].
template <int K>
__device__ __forceinline__ void wave_gemm(const _Float16* sArow, int lda,
                                          const _Float16* __restrict__ Wt,
                                          f8 acc[8]) {
  const int lane = threadIdx.x & 31;
  const int r = lane & 15;
  const int hi = lane >> 4;
  const _Float16* arow = sArow + r * lda;
#pragma unroll
  for (int kb = 0; kb < K; kb += 32) {
    h16 a = load_a_frag(arow + kb, hi);
#pragma unroll
    for (int nt = 0; nt < 8; ++nt) {
      h16 b = *(const h16*)(Wt + (size_t)(nt * 16 + r) * K + kb + 16 * hi);
      acc[nt] = __builtin_amdgcn_wmma_f32_16x16x32_f16(
          false, a, false, b, (short)0, acc[nt], false, false);
    }
  }
}

// --------------------------- weight prep -----------------------------------
// Wt[n*K + k] = (f16) W[k*128 + n]   (W is [K,128] row-major)
__global__ void __launch_bounds__(256) k_transpose(const float* __restrict__ W,
                                                   _Float16* __restrict__ Wt,
                                                   int K) {
  int i = blockIdx.x * 256 + threadIdx.x;
  if (i < K * HID) {
    int nn = i / K, kk = i - nn * K;
    Wt[i] = (_Float16)W[(size_t)kk * HID + nn];
  }
}

__global__ void __launch_bounds__(256) k_zero(float* __restrict__ p,
                                              long long n) {
  long long i = (long long)blockIdx.x * 256 + threadIdx.x;
  if (i < n) p[i] = 0.0f;
}

__global__ void __launch_bounds__(256) k_f32_to_f16(const float* __restrict__ a,
                                                    _Float16* __restrict__ b,
                                                    long long n) {
  long long i = (long long)blockIdx.x * 256 + threadIdx.x;
  if (i < n) b[i] = (_Float16)a[i];
}

// --------------------------- node initializer ------------------------------
// x = relu(nf @ W0 + b0) @ W1 + b1 ; hidden built scalar (K=4), 2nd GEMM WMMA
__global__ void __launch_bounds__(256) k_node_init(
    const float* __restrict__ nf, const float* __restrict__ W0,
    const float* __restrict__ b0, const _Float16* __restrict__ W1t,
    const float* __restrict__ b1, float* __restrict__ x,
    _Float16* __restrict__ xh, int n) {
  extern __shared__ _Float16 sA[];
  const int lda = HID + PAD;
  const int row0 = blockIdx.x * 128;
  for (int i = threadIdx.x; i < 128 * HID; i += 256) {
    int rr = i >> 7, cc = i & 127;
    int row = row0 + rr;
    float v = 0.0f;
    if (row < n) {
      const float* nfr = nf + (size_t)row * 4;
      v = fmaf(nfr[0], W0[0 * HID + cc],
          fmaf(nfr[1], W0[1 * HID + cc],
          fmaf(nfr[2], W0[2 * HID + cc],
          fmaf(nfr[3], W0[3 * HID + cc], b0[cc]))));
      v = fmaxf(v, 0.0f);
    }
    sA[rr * lda + cc] = (_Float16)v;
  }
  __syncthreads();
  const int wave = threadIdx.x >> 5;
  f8 acc[8] = {};
  wave_gemm<HID>(sA + wave * 16 * lda, lda, W1t, acc);
  const int lane = threadIdx.x & 31, r = lane & 15, hi = lane >> 4;
#pragma unroll
  for (int nt = 0; nt < 8; ++nt) {
    int col = nt * 16 + r;
    float bs = b1[col];
#pragma unroll
    for (int v = 0; v < 8; ++v) {
      int grow = row0 + wave * 16 + v + 8 * hi;
      if (grow < n) {
        float val = acc[nt][v] + bs;
        x[(size_t)grow * HID + col] = val;
        xh[(size_t)grow * HID + col] = (_Float16)val;
      }
    }
  }
}

// --------------------------- edge encoder ----------------------------------
// e = relu(ea * We0 + be0) @ We1 + be1  -> eh (f16)
__global__ void __launch_bounds__(256) k_edge_enc(
    const float* __restrict__ ea, const float* __restrict__ We0,
    const float* __restrict__ be0, const _Float16* __restrict__ We1t,
    const float* __restrict__ be1, _Float16* __restrict__ ehb, int e) {
  extern __shared__ _Float16 sA[];
  const int lda = HID + PAD;
  const int row0 = blockIdx.x * 128;
  for (int i = threadIdx.x; i < 128 * HID; i += 256) {
    int rr = i >> 7, cc = i & 127;
    int row = row0 + rr;
    float v = 0.0f;
    if (row < e) v = fmaxf(fmaf(ea[row], We0[cc], be0[cc]), 0.0f);
    sA[rr * lda + cc] = (_Float16)v;
  }
  __syncthreads();
  const int wave = threadIdx.x >> 5;
  f8 acc[8] = {};
  wave_gemm<HID>(sA + wave * 16 * lda, lda, We1t, acc);
  const int lane = threadIdx.x & 31, r = lane & 15, hi = lane >> 4;
#pragma unroll
  for (int nt = 0; nt < 8; ++nt) {
    int col = nt * 16 + r;
    float bs = be1[col];
#pragma unroll
    for (int v = 0; v < 8; ++v) {
      int grow = row0 + wave * 16 + v + 8 * hi;
      if (grow < e)
        ehb[(size_t)grow * HID + col] = (_Float16)(acc[nt][v] + bs);
    }
  }
}

// --------------------------- message hidden --------------------------------
// mh = relu(concat(x[dst], x[src], e) @ Wm0 + bm0)  (K=384, gathered into LDS)
__global__ void __launch_bounds__(128) k_msg_hidden(
    const _Float16* __restrict__ xh, const _Float16* __restrict__ ehb,
    const int* __restrict__ srcI, const int* __restrict__ dstI,
    const _Float16* __restrict__ Wm0t, const float* __restrict__ bm0,
    _Float16* __restrict__ mh, int e) {
  extern __shared__ _Float16 sA[];
  const int lda = 3 * HID + PAD;
  const int row0 = blockIdx.x * 64;
  for (int c = threadIdx.x; c < 64 * 48; c += 128) {
    int rr = c / 48, s = c % 48;
    int row = row0 + rr;
    int seg = s >> 4, off = (s & 15) * 8;
    h8 vv = {};
    if (row < e) {
      const _Float16* p;
      if (seg == 0)      p = xh + (size_t)dstI[row] * HID + off;
      else if (seg == 1) p = xh + (size_t)srcI[row] * HID + off;
      else               p = ehb + (size_t)row * HID + off;
      vv = *(const h8*)p;
    }
    *(h8*)(sA + rr * lda + seg * HID + off) = vv;
  }
  __syncthreads();
  const int wave = threadIdx.x >> 5;
  f8 acc[8] = {};
  wave_gemm<3 * HID>(sA + wave * 16 * lda, lda, Wm0t, acc);
  const int lane = threadIdx.x & 31, r = lane & 15, hi = lane >> 4;
#pragma unroll
  for (int nt = 0; nt < 8; ++nt) {
    int col = nt * 16 + r;
    float bs = bm0[col];
#pragma unroll
    for (int v = 0; v < 8; ++v) {
      int grow = row0 + wave * 16 + v + 8 * hi;
      if (grow < e)
        mh[(size_t)grow * HID + col] = (_Float16)fmaxf(acc[nt][v] + bs, 0.0f);
    }
  }
}

// --------------------------- message out + scatter -------------------------
// m = mh @ Wm1 + bm1 ; agg[dst] += m  (fused atomic scatter epilogue)
__global__ void __launch_bounds__(256) k_msg_out(
    const _Float16* __restrict__ mh, const _Float16* __restrict__ Wm1t,
    const float* __restrict__ bm1, const int* __restrict__ dstI,
    float* __restrict__ agg, int e) {
  extern __shared__ _Float16 sA[];
  const int lda = HID + PAD;
  const int row0 = blockIdx.x * 128;
  for (int c = threadIdx.x; c < 128 * 16; c += 256) {
    int rr = c >> 4, off = (c & 15) * 8;
    int row = row0 + rr;
    h8 vv = {};
    if (row < e) vv = *(const h8*)(mh + (size_t)row * HID + off);
    *(h8*)(sA + rr * lda + off) = vv;
  }
  __syncthreads();
  const int wave = threadIdx.x >> 5;
  f8 acc[8] = {};
  wave_gemm<HID>(sA + wave * 16 * lda, lda, Wm1t, acc);
  const int lane = threadIdx.x & 31, r = lane & 15, hi = lane >> 4;
  int drow[8];
#pragma unroll
  for (int v = 0; v < 8; ++v) {
    int grow = row0 + wave * 16 + v + 8 * hi;
    drow[v] = (grow < e) ? dstI[grow] : -1;
  }
#pragma unroll
  for (int nt = 0; nt < 8; ++nt) {
    int col = nt * 16 + r;
    float bs = bm1[col];
#pragma unroll
    for (int v = 0; v < 8; ++v)
      if (drow[v] >= 0)
        atomicAdd(agg + (size_t)drow[v] * HID + col, acc[nt][v] + bs);
  }
}

// --------------------------- update hidden ---------------------------------
// uh = relu(concat(x, agg) @ Wu0 + bu0)  (K=256)
__global__ void __launch_bounds__(128) k_upd_hidden(
    const _Float16* __restrict__ xh, const _Float16* __restrict__ aggh,
    const _Float16* __restrict__ Wu0t, const float* __restrict__ bu0,
    _Float16* __restrict__ uh, int n) {
  extern __shared__ _Float16 sA[];
  const int lda = 2 * HID + PAD;
  const int row0 = blockIdx.x * 64;
  for (int c = threadIdx.x; c < 64 * 32; c += 128) {
    int rr = c >> 5, s = c & 31;
    int row = row0 + rr;
    int seg = s >> 4, off = (s & 15) * 8;
    h8 vv = {};
    if (row < n) {
      const _Float16* p = (seg == 0) ? (xh + (size_t)row * HID + off)
                                     : (aggh + (size_t)row * HID + off);
      vv = *(const h8*)p;
    }
    *(h8*)(sA + rr * lda + seg * HID + off) = vv;
  }
  __syncthreads();
  const int wave = threadIdx.x >> 5;
  f8 acc[8] = {};
  wave_gemm<2 * HID>(sA + wave * 16 * lda, lda, Wu0t, acc);
  const int lane = threadIdx.x & 31, r = lane & 15, hi = lane >> 4;
#pragma unroll
  for (int nt = 0; nt < 8; ++nt) {
    int col = nt * 16 + r;
    float bs = bu0[col];
#pragma unroll
    for (int v = 0; v < 8; ++v) {
      int grow = row0 + wave * 16 + v + 8 * hi;
      if (grow < n)
        uh[(size_t)grow * HID + col] = (_Float16)fmaxf(acc[nt][v] + bs, 0.0f);
    }
  }
}

// --------------------------- update out ------------------------------------
// x = uh @ Wu1 + bu1  (writes f32 master + f16 copy in place)
__global__ void __launch_bounds__(256) k_upd_out(
    const _Float16* __restrict__ uh, const _Float16* __restrict__ Wu1t,
    const float* __restrict__ bu1, float* __restrict__ x,
    _Float16* __restrict__ xh, int n) {
  extern __shared__ _Float16 sA[];
  const int lda = HID + PAD;
  const int row0 = blockIdx.x * 128;
  for (int c = threadIdx.x; c < 128 * 16; c += 256) {
    int rr = c >> 4, off = (c & 15) * 8;
    int row = row0 + rr;
    h8 vv = {};
    if (row < n) vv = *(const h8*)(uh + (size_t)row * HID + off);
    *(h8*)(sA + rr * lda + off) = vv;
  }
  __syncthreads();
  const int wave = threadIdx.x >> 5;
  f8 acc[8] = {};
  wave_gemm<HID>(sA + wave * 16 * lda, lda, Wu1t, acc);
  const int lane = threadIdx.x & 31, r = lane & 15, hi = lane >> 4;
#pragma unroll
  for (int nt = 0; nt < 8; ++nt) {
    int col = nt * 16 + r;
    float bs = bu1[col];
#pragma unroll
    for (int v = 0; v < 8; ++v) {
      int grow = row0 + wave * 16 + v + 8 * hi;
      if (grow < n) {
        float val = acc[nt][v] + bs;
        x[(size_t)grow * HID + col] = val;
        xh[(size_t)grow * HID + col] = (_Float16)val;
      }
    }
  }
}

// --------------------------- pooling + head --------------------------------
__global__ void __launch_bounds__(256) k_pool(const float* __restrict__ x,
                                              const int* __restrict__ batch,
                                              float* __restrict__ sums,
                                              float* __restrict__ cnts,
                                              int n) {
  long long i = (long long)blockIdx.x * 256 + threadIdx.x;
  if (i < (long long)n * HID) {
    int node = (int)(i >> 7), c = (int)(i & 127);
    int b = batch[node];
    atomicAdd(sums + (size_t)b * HID + c, x[i]);
    if (c == 0) atomicAdd(cnts + b, 1.0f);
  }
}

__global__ void __launch_bounds__(256) k_final(
    const float* __restrict__ sums, const float* __restrict__ cnts,
    const float* __restrict__ W0, const float* __restrict__ b0,
    const float* __restrict__ W1, const float* __restrict__ b1,
    float* __restrict__ out) {
  __shared__ float P[16 * HID];
  __shared__ float Hd[16 * HID];
  for (int i = threadIdx.x; i < 16 * HID; i += 256) {
    int rr = i >> 7;
    P[i] = sums[i] / fmaxf(cnts[rr], 1.0f);
  }
  __syncthreads();
  for (int i = threadIdx.x; i < 16 * HID; i += 256) {
    int rr = i >> 7, cc = i & 127;
    float a = b0[cc];
    for (int k = 0; k < HID; ++k) a = fmaf(P[rr * HID + k], W0[k * HID + cc], a);
    Hd[i] = fmaxf(a, 0.0f);
  }
  __syncthreads();
  for (int i = threadIdx.x; i < 16 * HID; i += 256) {
    int rr = i >> 7, cc = i & 127;
    float a = b1[cc];
    for (int k = 0; k < HID; ++k) a = fmaf(Hd[rr * HID + k], W1[k * HID + cc], a);
    out[i] = a;
  }
}

// ---------------------------------------------------------------------------
extern "C" void kernel_launch(void* const* d_in, const int* in_sizes, int n_in,
                              void* d_out, int out_size, void* d_ws,
                              size_t ws_size, hipStream_t stream) {
  const float* nf   = (const float*)d_in[0];
  const float* ea   = (const float*)d_in[1];
  const float* niW0 = (const float*)d_in[2];
  const float* nib0 = (const float*)d_in[3];
  const float* niW1 = (const float*)d_in[4];
  const float* nib1 = (const float*)d_in[5];
  const float* We0  = (const float*)d_in[6];
  const float* be0  = (const float*)d_in[7];
  const float* We1  = (const float*)d_in[8];
  const float* be1  = (const float*)d_in[9];
  const float* Wm0  = (const float*)d_in[10];
  const float* bm0  = (const float*)d_in[11];
  const float* Wm1  = (const float*)d_in[12];
  const float* bm1  = (const float*)d_in[13];
  const float* Wu0  = (const float*)d_in[14];
  const float* bu0  = (const float*)d_in[15];
  const float* Wu1  = (const float*)d_in[16];
  const float* bu1  = (const float*)d_in[17];
  const float* pW0  = (const float*)d_in[18];
  const float* pb0  = (const float*)d_in[19];
  const float* pW1  = (const float*)d_in[20];
  const float* pb1  = (const float*)d_in[21];
  const int* eidx   = (const int*)d_in[22];
  const int* batch  = (const int*)d_in[23];

  const int N = in_sizes[0] / 4;
  const int E = in_sizes[1];
  const int B = 16;
  const int* srcI = eidx;      // edge_index[0] = source
  const int* dstI = eidx + E;  // edge_index[1] = destination

  (void)n_in; (void)out_size; (void)ws_size;

  // workspace carve (256-byte aligned)
  char* ws = (char*)d_ws;
  size_t off = 0;
  auto carve = [&](size_t bytes) -> void* {
    void* p = ws + off;
    off = (off + bytes + 255) & ~(size_t)255;
    return p;
  };
  _Float16* W1t  = (_Float16*)carve((size_t)HID * HID * 2);
  _Float16* We1t = (_Float16*)carve((size_t)NL * HID * HID * 2);
  _Float16* Wm0t = (_Float16*)carve((size_t)NL * 3 * HID * HID * 2);
  _Float16* Wm1t = (_Float16*)carve((size_t)NL * HID * HID * 2);
  _Float16* Wu0t = (_Float16*)carve((size_t)NL * 2 * HID * HID * 2);
  _Float16* Wu1t = (_Float16*)carve((size_t)NL * HID * HID * 2);
  float*    x    = (float*)carve((size_t)N * HID * 4);
  _Float16* xh   = (_Float16*)carve((size_t)N * HID * 2);
  _Float16* ehb  = (_Float16*)carve((size_t)E * HID * 2);
  _Float16* mh   = (_Float16*)carve((size_t)E * HID * 2);
  float*    agg  = (float*)carve((size_t)N * HID * 4);
  _Float16* aggh = (_Float16*)carve((size_t)N * HID * 2);
  _Float16* uh   = (_Float16*)carve((size_t)N * HID * 2);
  float*    sums = (float*)carve((size_t)B * HID * 4 + (size_t)B * 4);
  float*    cnts = sums + B * HID;

  const long long NH = (long long)N * HID;
  const int gW1 = (HID * HID + 255) / 256;

  // ---- weight prep (transpose + f16 convert) ----
  k_transpose<<<gW1, 256, 0, stream>>>(niW1, W1t, HID);
  for (int l = 0; l < NL; ++l) {
    k_transpose<<<gW1, 256, 0, stream>>>(We1 + (size_t)l * HID * HID,
                                         We1t + (size_t)l * HID * HID, HID);
    k_transpose<<<(3 * HID * HID + 255) / 256, 256, 0, stream>>>(
        Wm0 + (size_t)l * 3 * HID * HID, Wm0t + (size_t)l * 3 * HID * HID,
        3 * HID);
    k_transpose<<<gW1, 256, 0, stream>>>(Wm1 + (size_t)l * HID * HID,
                                         Wm1t + (size_t)l * HID * HID, HID);
    k_transpose<<<(2 * HID * HID + 255) / 256, 256, 0, stream>>>(
        Wu0 + (size_t)l * 2 * HID * HID, Wu0t + (size_t)l * 2 * HID * HID,
        2 * HID);
    k_transpose<<<gW1, 256, 0, stream>>>(Wu1 + (size_t)l * HID * HID,
                                         Wu1t + (size_t)l * HID * HID, HID);
  }

  const size_t lds128 = (size_t)128 * (HID + PAD) * 2;       // 34.0 KB
  const size_t lds384 = (size_t)64 * (3 * HID + PAD) * 2;    // 49.0 KB
  const size_t lds256 = (size_t)64 * (2 * HID + PAD) * 2;    // 33.0 KB

  // ---- node initializer ----
  k_node_init<<<(N + 127) / 128, 256, lds128, stream>>>(nf, niW0, nib0, W1t,
                                                        nib1, x, xh, N);

  // ---- message-passing layers ----
  for (int l = 0; l < NL; ++l) {
    k_edge_enc<<<(E + 127) / 128, 256, lds128, stream>>>(
        ea, We0 + (size_t)l * HID, be0 + (size_t)l * HID,
        We1t + (size_t)l * HID * HID, be1 + (size_t)l * HID, ehb, E);
    k_msg_hidden<<<(E + 63) / 64, 128, lds384, stream>>>(
        xh, ehb, srcI, dstI, Wm0t + (size_t)l * 3 * HID * HID,
        bm0 + (size_t)l * HID, mh, E);
    k_zero<<<(int)((NH + 255) / 256), 256, 0, stream>>>(agg, NH);
    k_msg_out<<<(E + 127) / 128, 256, lds128, stream>>>(
        mh, Wm1t + (size_t)l * HID * HID, bm1 + (size_t)l * HID, dstI, agg, E);
    k_f32_to_f16<<<(int)((NH + 255) / 256), 256, 0, stream>>>(agg, aggh, NH);
    k_upd_hidden<<<(N + 63) / 64, 128, lds256, stream>>>(
        xh, aggh, Wu0t + (size_t)l * 2 * HID * HID, bu0 + (size_t)l * HID, uh,
        N);
    k_upd_out<<<(N + 127) / 128, 256, lds128, stream>>>(
        uh, Wu1t + (size_t)l * HID * HID, bu1 + (size_t)l * HID, x, xh, N);
  }

  // ---- per-graph mean pooling + head MLP ----
  k_zero<<<1, 256, 0, stream>>>(sums, (long long)B * HID + B);
  k_pool<<<(int)((NH + 255) / 256), 256, 0, stream>>>(x, batch, sums, cnts, N);
  k_final<<<1, 256, 0, stream>>>(sums, cnts, pW0, pb0, pW1, pb1, (float*)d_out);
}
